// MHSA_33131377722100
// MI455X (gfx1250) — matrix operand order repaired and checked by reading, not compiled, not run on hardware
//
#include <hip/hip_runtime.h>

#define D_MODEL   1024
#define NUM_HEADS 16
#define HEAD_W    64
#define SEQ       2048
#define BATCH     2

typedef __attribute__((ext_vector_type(16))) _Float16 v16h;
typedef __attribute__((ext_vector_type(8)))  _Float16 v8h;
typedef __attribute__((ext_vector_type(8)))  float    v8f;
typedef __attribute__((ext_vector_type(4)))  int      v4i;

union H16 { v16h v; v8h h[2]; _Float16 e[16]; };

// ---- CDNA5 async global->LDS path (guarded; falls back to sync copy) ------
#if defined(__has_builtin)
#  if __has_builtin(__builtin_amdgcn_global_load_async_to_lds_b128) && \
      __has_builtin(__builtin_amdgcn_s_wait_asynccnt)
#    define USE_ASYNC_LDS 1
#  endif
#endif
#ifndef USE_ASYNC_LDS
#  define USE_ASYNC_LDS 0
#endif

typedef __attribute__((address_space(1))) v4i g_v4i;   // global (__device__)
typedef __attribute__((address_space(3))) v4i l_v4i;   // LDS

static __device__ inline void copy16_g2l(_Float16* ldst, const _Float16* gsrc) {
#if USE_ASYNC_LDS
  // AS casts routed through integers: generic global == AS1 numerically;
  // low 32 bits of a generic LDS pointer are the LDS byte address (AS3).
  g_v4i* gp = (g_v4i*)(unsigned long long)gsrc;
  l_v4i* lp = (l_v4i*)(unsigned int)(unsigned long long)ldst;
  __builtin_amdgcn_global_load_async_to_lds_b128(gp, lp, 0, 0);
#else
  *(v8h*)ldst = *(const v8h*)gsrc;
#endif
}
static __device__ inline void async_wait() {
#if USE_ASYNC_LDS
  __builtin_amdgcn_s_wait_asynccnt(0);
#endif
}

// B-fragment: 16 contiguous halves at p (per-lane column of B, K ascending)
static __device__ inline v16h load_b_frag(const _Float16* p) {
  H16 u;
  u.h[0] = *(const v8h*)(p);
  u.h[1] = *(const v8h*)(p + 8);
  return u.v;
}
// A-fragment: 8 halves at p, 8 halves at p+16 (per-lane row of A, split K)
static __device__ inline v16h load_a_frag(const _Float16* p) {
  H16 u;
  u.h[0] = *(const v8h*)(p);
  u.h[1] = *(const v8h*)(p + 16);
  return u.v;
}

static __device__ inline v8f wmma_f16(v16h a, v16h b, v8f c) {
  return __builtin_amdgcn_wmma_f32_16x16x32_f16(false, a, false, b, (short)0, c,
                                                false, false);
}

// ---------------------------------------------------------------------------
// Stage 0: f32 -> f16 conversion (weights)
// ---------------------------------------------------------------------------
__global__ void cvt_f32_f16(const float* __restrict__ src,
                            _Float16* __restrict__ dst, int n) {
  int i = blockIdx.x * blockDim.x + threadIdx.x;
  if (i < n) dst[i] = (_Float16)src[i];
}

// ---------------------------------------------------------------------------
// Stage 1: QKV projection.  out = X @ W^T  (nn.Linear): both operands have K
// contiguous -> direct WMMA fragment loads, no transpose.  One wave computes
// a 32(M) x 64(N) tile.  MODE 0: store [bh,t,d] (Q/K, scale folds 1/8 for Q).
// MODE 1: store V^T [bh,d,t] via LDS transpose for coalesced 16B stores.
// ---------------------------------------------------------------------------
template <int MODE>
__global__ __launch_bounds__(32)
void qkv_proj(const float* __restrict__ X, const _Float16* __restrict__ W,
              _Float16* __restrict__ out, float scale) {
  const int lane = threadIdx.x & 31;
  const int la = lane & 15, hi = lane >> 4;
  const int Nbase = blockIdx.x * 64;
  const int Mbase = blockIdx.y * 32;

  v8f acc[2][4] = {};

  for (int k = 0; k < D_MODEL; k += 32) {
    v16h a[2];
#pragma unroll
    for (int mt = 0; mt < 2; ++mt) {
      const float* p = X + (size_t)(Mbase + mt * 16 + la) * D_MODEL + k + hi * 8;
      H16 u;
#pragma unroll
      for (int i = 0; i < 8; ++i) u.e[i]     = (_Float16)p[i];
#pragma unroll
      for (int i = 0; i < 8; ++i) u.e[8 + i] = (_Float16)p[16 + i];
      a[mt] = u.v;
    }
#pragma unroll
    for (int nt = 0; nt < 4; ++nt) {
      v16h b = load_a_frag(W + (size_t)(Nbase + nt * 16 + la) * D_MODEL + k + hi * 8);
#pragma unroll
      for (int mt = 0; mt < 2; ++mt)
        acc[mt][nt] = wmma_f16(a[mt], b, acc[mt][nt]);
    }
  }

  const int bb = Mbase >> 11;            // batch (tile never straddles batch)
  const int t0 = Mbase & (SEQ - 1);

  if constexpr (MODE == 0) {
    // Q/K: [bh, t, d] layout, per-column contiguous base hoisted
#pragma unroll
    for (int nt = 0; nt < 4; ++nt) {
      int col = Nbase + nt * 16 + la;
      int h = col >> 6, d = col & (HEAD_W - 1);
      _Float16* base =
          out + ((size_t)(bb * NUM_HEADS + h) * SEQ + t0) * HEAD_W + d;
#pragma unroll
      for (int mt = 0; mt < 2; ++mt)
#pragma unroll
        for (int j = 0; j < 8; ++j)
          base[(size_t)(mt * 16 + hi * 8 + j) * HEAD_W] =
              (_Float16)(acc[mt][nt][j] * scale);
    }
  } else {
    // V^T: transpose 32x64 C tile through LDS, then coalesced 16B row stores
    __shared__ __align__(32) _Float16 Tt[64][32];
#pragma unroll
    for (int nt = 0; nt < 4; ++nt)
#pragma unroll
      for (int mt = 0; mt < 2; ++mt)
#pragma unroll
        for (int j = 0; j < 8; ++j)
          Tt[nt * 16 + la][mt * 16 + hi * 8 + j] = (_Float16)acc[mt][nt][j];
    __syncthreads();   // single wave: DS ops are in-order; fence vs reorder
#pragma unroll
    for (int r = 0; r < 8; ++r) {
      int i   = r * 32 + lane;   // 256 chunks of 8 halves
      int row = i >> 2;          // 0..63  (local output feature)
      int ch  = i & 3;           // 0..3   (8-half chunk along t)
      int col = Nbase + row;
      int h = col >> 6, d = col & (HEAD_W - 1);
      _Float16* dst = out +
          ((size_t)(bb * NUM_HEADS + h) * HEAD_W + d) * SEQ + t0 + ch * 8;
      *(v8h*)dst = *(const v8h*)&Tt[row][ch * 8];
    }
  }
}

// ---------------------------------------------------------------------------
// Stage 2: flash attention (transposed formulation) with ALiBi + causal mask.
// Block = 8 waves, 128 queries of one (b,h).  Double-buffered 32-key K/V^T
// tiles staged via async global->LDS (prefetch kb+1 behind WMMA on kb).
// Per wave/key-block: S^T = K Q^T (4 WMMA), per-lane online softmax (queries
// live in lanes -> scalar m/l, one shfl_xor(16) per reduction), P^T packed
// in-register via cvt_pk + half-wave swap, O^T += V^T P^T (4 WMMA).
// ---------------------------------------------------------------------------
__global__ __launch_bounds__(256)
void attn(const _Float16* __restrict__ Qh, const _Float16* __restrict__ Kh,
          const _Float16* __restrict__ VT, float* __restrict__ out) {
  __shared__ __align__(32) _Float16 Kt2[2][32 * 64];   // [buf][key][dim]
  __shared__ __align__(32) _Float16 Vt2[2][64 * 32];   // [buf][dim][key]

  const int tid  = threadIdx.x;
  const int w    = tid >> 5;
  const int lane = tid & 31;
  const int la = lane & 15, hi = lane >> 4;

  const int bh = blockIdx.y;                 // b*H + h
  const int h  = bh & (NUM_HEADS - 1);
  const int b  = bh >> 4;
  const int qgrp  = blockIdx.x * 128;
  const int qbase = qgrp + w * 16;
  const float slope = exp2f(-0.5f * (float)(h + 1));   // 2^(-(h+1)*8/H)

  const size_t bhS = (size_t)bh * SEQ;

  // staging indices (256 threads x 16B per tile)
  const int sidx = tid * 8;
  const int skey = sidx >> 6, sdim = sidx & 63;
  const int sd2  = sidx >> 5, sk2  = sidx & 31;

  // Q as B-operand: column = this lane's query, K = head dims (contiguous)
  v16h bq0, bq1;
  {
    const _Float16* qrow = Qh + (bhS + qbase + la) * HEAD_W;
    bq0 = load_a_frag(qrow + hi * 8);         // dims  0..31
    bq1 = load_a_frag(qrow + 32 + hi * 8);    // dims 32..63
  }

  const int qi = qbase + la;     // this lane's query index
  float m = -3.0e38f, l = 0.0f;  // per-query softmax state (per-lane scalars)
  v8f O[4] = {};                 // O^T: [dim tile][8 consecutive dims]

  const int nkb = blockIdx.x * 4 + 4;   // key blocks up to group diagonal

  // prologue: stage block 0 into buffer 0
  copy16_g2l(Kt2[0] + skey * 64 + sdim, Kh + (bhS + skey) * HEAD_W + sdim);
  copy16_g2l(Vt2[0] + sd2 * 32 + sk2,
             VT + ((size_t)bh * HEAD_W + sd2) * SEQ + sk2);
  async_wait();
  __syncthreads();

  for (int kb = 0; kb < nkb; ++kb) {
    const int jbase = kb * 32;
    const int cur = kb & 1;

    // prefetch next key block into the other buffer (async, no wait yet)
    if (kb + 1 < nkb) {
      const int jn = jbase + 32;
      copy16_g2l(Kt2[1 - cur] + skey * 64 + sdim,
                 Kh + (bhS + jn + skey) * HEAD_W + sdim);
      copy16_g2l(Vt2[1 - cur] + sd2 * 32 + sk2,
                 VT + ((size_t)bh * HEAD_W + sd2) * SEQ + jn + sk2);
    }

    // wave-uniform causal skip, forced scalar so EXEC is untouched
    if (__builtin_amdgcn_readfirstlane((jbase < qbase + 16) ? 1 : 0)) {
      const _Float16* Kt = Kt2[cur];
      const _Float16* Vt = Vt2[cur];

      // ---- S^T = K Q^T : two 16(key) x 16(query) C tiles ----
      v8f st0 = {}, st1 = {};
      {
        v16h a00 = load_a_frag(Kt + la * 64 + hi * 8);         // keys 0-15
        v16h a01 = load_a_frag(Kt + la * 64 + 32 + hi * 8);
        st0 = wmma_f16(a00, bq0, st0);
        st0 = wmma_f16(a01, bq1, st0);
        v16h a10 = load_a_frag(Kt + (16 + la) * 64 + hi * 8);  // keys 16-31
        v16h a11 = load_a_frag(Kt + (16 + la) * 64 + 32 + hi * 8);
        st1 = wmma_f16(a10, bq0, st1);
        st1 = wmma_f16(a11, bq1, st1);
      }

      // ---- ALiBi + causal mask; per-lane row max over 16 own keys ----
      float p0[8], p1[8];
      float r = -3.0e38f;
#pragma unroll
      for (int j = 0; j < 8; ++j) {
        int k0 = jbase + hi * 8 + j;        // key of st0[j] (this lane)
        int k1 = k0 + 16;                   // key of st1[j]
        float v0 = st0[j] + slope * (float)(k0 - qi) + ((k0 > qi) ? -1e9f : 0.0f);
        float v1 = st1[j] + slope * (float)(k1 - qi) + ((k1 > qi) ? -1e9f : 0.0f);
        p0[j] = v0; p1[j] = v1;
        r = fmaxf(r, fmaxf(v0, v1));
      }
      r = fmaxf(r, __shfl_xor(r, 16));      // combine the two key halves
      float mn = fmaxf(m, r);
      float al = __expf(m - mn);
      m = mn;
      float rs = 0.0f;
#pragma unroll
      for (int j = 0; j < 8; ++j) {
        p0[j] = __expf(p0[j] - mn);
        p1[j] = __expf(p1[j] - mn);
        rs += p0[j] + p1[j];
      }
      rs += __shfl_xor(rs, 16);
      l = l * al + rs;
#pragma unroll
      for (int dt = 0; dt < 4; ++dt) O[dt] *= al;

      // ---- pack P^T as B-operand: cvt_pk pairs + half-wave swap ----
      union PKU { float f; _Float16 e[2]; };
      union BFU { v16h v; float f[8]; };
      float pk0[4], pk1[4];
#pragma unroll
      for (int v = 0; v < 4; ++v) {
        PKU a; a.e[0] = (_Float16)p0[2 * v]; a.e[1] = (_Float16)p0[2 * v + 1];
        pk0[v] = a.f;
        PKU c; c.e[0] = (_Float16)p1[2 * v]; c.e[1] = (_Float16)p1[2 * v + 1];
        pk1[v] = c.f;
      }
      BFU bp;
#pragma unroll
      for (int v = 0; v < 4; ++v) {
        float t0 = __shfl_xor(pk0[v], 16);  // partner half's tile-0 packs
        float t1 = __shfl_xor(pk1[v], 16);  // partner half's tile-1 packs
        bp.f[v]     = hi ? t1 : pk0[v];     // keys 2v..2v+1   / 16+2v..
        bp.f[4 + v] = hi ? pk1[v] : t0;     // keys 8+2v..     / 24+2v..
      }

      // ---- O^T += V^T P^T ----
#pragma unroll
      for (int dt = 0; dt < 4; ++dt) {
        v16h av = load_b_frag(Vt + (dt * 16 + la) * 32 + hi * 16);
        O[dt] = wmma_f16(av, bp.v, O[dt]);
      }
    }
    async_wait();      // prefetch of kb+1 complete before anyone reads it
    __syncthreads();
  }

  // ---- epilogue: out[b, t=qi, h*64 + dim] = O^T / l, packed 32B stores ----
  {
    float inv = 1.0f / l;
    float* orow = out + ((size_t)b * SEQ + qi) * D_MODEL + h * HEAD_W + hi * 8;
#pragma unroll
    for (int dt = 0; dt < 4; ++dt) {
      v8f rr = O[dt] * inv;
      *(v8f*)(orow + dt * 16) = rr;
    }
  }
}

// ---------------------------------------------------------------------------
extern "C" void kernel_launch(void* const* d_in, const int* in_sizes, int n_in,
                              void* d_out, int out_size, void* d_ws,
                              size_t ws_size, hipStream_t stream) {
  const float* q  = (const float*)d_in[0];
  const float* k  = (const float*)d_in[1];
  const float* v  = (const float*)d_in[2];
  // d_in[3] = additive causal mask (recomputed analytically in-kernel)
  const float* Wq = (const float*)d_in[4];
  const float* Wk = (const float*)d_in[5];
  const float* Wv = (const float*)d_in[6];
  float* out = (float*)d_out;

  const size_t WELEM = (size_t)D_MODEL * D_MODEL;                  // 1M
  const size_t QELEM = (size_t)BATCH * NUM_HEADS * SEQ * HEAD_W;   // 4M
  _Float16* Wqh = (_Float16*)d_ws;
  _Float16* Wkh = Wqh + WELEM;
  _Float16* Wvh = Wkh + WELEM;
  _Float16* Qh  = Wvh + WELEM;
  _Float16* Kh  = Qh + QELEM;
  _Float16* VT  = Kh + QELEM;

  {
    int n = (int)WELEM;
    dim3 cg((n + 255) / 256);
    cvt_f32_f16<<<cg, 256, 0, stream>>>(Wq, Wqh, n);
    cvt_f32_f16<<<cg, 256, 0, stream>>>(Wk, Wkh, n);
    cvt_f32_f16<<<cg, 256, 0, stream>>>(Wv, Wvh, n);
  }

  dim3 pg(D_MODEL / 64, (BATCH * SEQ) / 32);
  qkv_proj<0><<<pg, 32, 0, stream>>>(q, Wqh, Qh, 0.125f);  // fold 1/sqrt(64)
  qkv_proj<0><<<pg, 32, 0, stream>>>(k, Wkh, Kh, 1.0f);
  qkv_proj<1><<<pg, 32, 0, stream>>>(v, Wvh, VT, 1.0f);    // store V^T

  dim3 ag(SEQ / 128, BATCH * NUM_HEADS);
  attn<<<ag, 256, 0, stream>>>(Qh, Kh, VT, out);
}
